// MultiHeadAttention_79594333929981
// MI455X (gfx1250) — matrix operand order repaired and checked by reading, not compiled, not run on hardware
//
#include <hip/hip_runtime.h>
#include <hip/hip_bf16.h>

// ---------------------------------------------------------------------------
// MI455X (gfx1250) fused multi-head attention, wave32 + v_wmma_f32_16x16x32_bf16
// B=2, S=4096, D=512, H=8, DK=64.  Softmax is over the HEADS axis (faithful to
// the reference bug) -> pointwise per (q,k), so the key loop fuses trivially
// and the causal mask cancels into uniform 1/8 rows exactly as fp32 JAX does.
// Scores never touch HBM (saves ~2 GiB of traffic @ 23.3 TB/s).
// ---------------------------------------------------------------------------

typedef __bf16 bf16;
typedef __attribute__((ext_vector_type(16))) __bf16 bf16x16;
typedef __attribute__((ext_vector_type(8)))  __bf16 bf16x8;
typedef __attribute__((ext_vector_type(8)))  float  f32x8;
typedef __attribute__((ext_vector_type(4)))  float  f32x4;

#define WMMA_BF16(a, b, c) \
  __builtin_amdgcn_wmma_f32_16x16x32_bf16(false, (a), false, (b), (short)0, (c), false, false)

__device__ __forceinline__ bf16x16 load16(const bf16* p) { return *(const bf16x16*)p; }
__device__ __forceinline__ bf16x8  load8 (const bf16* p) { return *(const bf16x8*)p;  }
__device__ __forceinline__ bf16x16 cat8(bf16x8 lo, bf16x8 hi) {
  return __builtin_shufflevector(lo, hi, 0,1,2,3,4,5,6,7,8,9,10,11,12,13,14,15);
}

constexpr int Bz = 2, Sq = 4096, Dm = 512, Hn = 8, DK = 64;
constexpr int Mrows = Bz * Sq;            // 8192 flattened (b,s) rows

// ---------------------------------------------------------------------------
// fp32 -> bf16, 8 elements/thread (2x b128 in, 1x b128 out)
// ---------------------------------------------------------------------------
__global__ void cvt_f32_bf16(const float* __restrict__ in, bf16* __restrict__ out, int n8) {
  int i = blockIdx.x * 256 + threadIdx.x;
  if (i < n8) {
    f32x4 a = *(const f32x4*)(in + (size_t)i * 8);
    f32x4 b = *(const f32x4*)(in + (size_t)i * 8 + 4);
    bf16x8 o;
#pragma unroll
    for (int j = 0; j < 4; ++j) { o[j] = (bf16)a[j]; o[j + 4] = (bf16)b[j]; }
    *(bf16x8*)(out + (size_t)i * 8) = o;
  }
}

// ---------------------------------------------------------------------------
// QKV projection: y = x @ W.T.  Register-blocked: each wave computes a
// 32(M) x 64(N) strip -> 2 A-frags reused across 4 B-frags, 8 independent
// accumulator chains, 1.5 vmem per WMMA.
// grid = (Mrows/128, Dm/64, 3), block = 128 (4 waves, consecutive M strips).
// z selects Wq/Wk/Wv and the output layout:
//   z=0 -> Q [B,H,S,DK]   z=1 -> K [B,H,S,DK]   z=2 -> V^T [B,H,DK,S]
// ---------------------------------------------------------------------------
__global__ __launch_bounds__(128) void qkv_gemm(
    const bf16* __restrict__ xb,
    const bf16* __restrict__ wq, const bf16* __restrict__ wk, const bf16* __restrict__ wv,
    bf16* __restrict__ qh, bf16* __restrict__ kh, bf16* __restrict__ vt) {
  const int tid = threadIdx.x, wave = tid >> 5, lane = tid & 31;
  const int m = lane & 15, half = lane >> 4, n = m;
  const int row0 = blockIdx.x * 128 + wave * 32;
  const int n0   = blockIdx.y * 64;
  const int z    = blockIdx.z;

  const bf16* w = (z == 0) ? wq : (z == 1) ? wk : wv;
  const bf16* arow0 = xb + (size_t)(row0 + m) * Dm;        // M-tile 0
  const bf16* arow1 = arow0 + (size_t)16 * Dm;             // M-tile 1
  const bf16* brow  = w + (size_t)(n0 + n) * Dm;

  f32x8 acc[2][4] = {{{}, {}, {}, {}}, {{}, {}, {}, {}}};
#pragma unroll 4
  for (int kc = 0; kc < Dm / 32; ++kc) {
    const int d0 = kc * 32;
    // A 16x32 layout: lane m holds row m; K = half*16 + j (contiguous)
    bf16x16 a0 = load16(arow0 + d0 + half * 16);
    bf16x16 a1 = load16(arow1 + d0 + half * 16);
#pragma unroll
    for (int nt = 0; nt < 4; ++nt) {
      // B 32x16 layout: lane n holds column n; K = half*16 + j (contiguous)
      bf16x16 b = load16(brow + (size_t)nt * 16 * Dm + d0 + half * 16);
      acc[0][nt] = WMMA_BF16(a0, b, acc[0][nt]);
      acc[1][nt] = WMMA_BF16(a1, b, acc[1][nt]);
    }
  }

#pragma unroll
  for (int mt = 0; mt < 2; ++mt) {
#pragma unroll
    for (int nt = 0; nt < 4; ++nt) {
#pragma unroll
      for (int r = 0; r < 8; ++r) {
        const int grow = row0 + mt * 16 + r + 8 * half;    // 0..8191
        const int b    = grow >> 12;
        const int s    = grow & (Sq - 1);
        const int col  = n0 + nt * 16 + n;                 // 0..511
        const int h    = col >> 6, dk = col & 63;
        const bf16 v = (bf16)acc[mt][nt][r];
        if (z == 0)      qh[(((size_t)b * Hn + h) * Sq + s) * DK + dk] = v;
        else if (z == 1) kh[(((size_t)b * Hn + h) * Sq + s) * DK + dk] = v;
        else             vt[(((size_t)b * Hn + h) * DK + dk) * Sq + s] = v;
      }
    }
  }
}

// ---------------------------------------------------------------------------
// Fused attention.  grid = (Sq/16, B), block = 256 (8 waves; wave == head).
// 64-key chunks: 8 score WMMAs -> mask+scale -> LDS -> softmax over HEADS
// (pointwise per (q,k)) -> 2 attn A-frags -> 8 attn @ V WMMAs.  3 barriers
// per 16 WMMAs; LDS 32 KB of 320 KB so occupancy is VGPR-limited, not LDS.
// ---------------------------------------------------------------------------
__global__ __launch_bounds__(256) void attn_fused(
    const bf16* __restrict__ qh, const bf16* __restrict__ kh,
    const bf16* __restrict__ vt, bf16* __restrict__ ob) {
  __shared__ float sc[Hn][16][64];                 // 32 KB raw/normalized scores

  const int tid = threadIdx.x, h = tid >> 5, lane = tid & 31;
  const int m = lane & 15, half = lane >> 4, n = m;
  const int b  = blockIdx.y;
  const int q0 = blockIdx.x * 16;

  // Q A-fragments for contraction chunks d=[0,32) and d=[32,64)
  const bf16* qrow = qh + (((size_t)b * Hn + h) * Sq + q0 + m) * DK;
  const bf16x16 aQ0 = load16(qrow + half * 16);
  const bf16x16 aQ1 = load16(qrow + 32 + half * 16);

  const bf16* kbase = kh + ((size_t)b * Hn + h) * (size_t)Sq * DK;
  const bf16* vbase = vt + ((size_t)b * Hn + h) * (size_t)DK * Sq;

  f32x8 acc[4] = {{}, {}, {}, {}};                 // 16q x 64dk accumulator

  for (int k0 = 0; k0 < Sq; k0 += 64) {
    if (k0 + 64 < Sq)                              // pull next chunk toward L0
      __builtin_prefetch(kbase + (size_t)(k0 + 64 + n) * DK, 0, 1);

    // ---- scores: four 16-key tiles, contraction DK=64 (2 chained WMMAs each)
#pragma unroll
    for (int t = 0; t < 4; ++t) {
      const bf16* kr = kbase + (size_t)(k0 + t * 16 + n) * DK;
      bf16x16 bk0 = load16(kr + half * 16);        // d in [0,32)
      bf16x16 bk1 = load16(kr + 32 + half * 16);   // d in [32,64)
      f32x8 s = {};
      s = WMMA_BF16(aQ0, bk0, s);
      s = WMMA_BF16(aQ1, bk1, s);
      // mask BEFORE scaling by 1/sqrt(DK) (faithful to reference order)
#pragma unroll
      for (int r = 0; r < 8; ++r) {
        const int mm = r + 8 * half;
        const int gq = q0 + mm, gk = k0 + t * 16 + n;
        float v = s[r] + ((gk > gq) ? -1.0e9f : 0.0f);
        sc[h][mm][t * 16 + n] = v * 0.125f;        // 1/sqrt(64)
      }
    }
    __syncthreads();

    // ---- softmax across the 8 heads, pointwise per (q,k); in-place normalize
    for (int p = tid; p < 16 * 64; p += 256) {
      const int qq = p >> 6, kk = p & 63;
      float vv[8], mx = -3.0e38f;
#pragma unroll
      for (int hh = 0; hh < 8; ++hh) { vv[hh] = sc[hh][qq][kk]; mx = fmaxf(mx, vv[hh]); }
      float sum = 0.0f;
#pragma unroll
      for (int hh = 0; hh < 8; ++hh) { vv[hh] = __expf(vv[hh] - mx); sum += vv[hh]; }
      const float r = __builtin_amdgcn_rcpf(sum);
#pragma unroll
      for (int hh = 0; hh < 8; ++hh) sc[hh][qq][kk] = vv[hh] * r;
    }
    __syncthreads();

    // ---- attn A-fragments for the two 32-key sub-chunks (contiguous K = half*16 + j)
    bf16x16 aP0, aP1;
#pragma unroll
    for (int j = 0; j < 16; ++j) {
      const int kk = half * 16 + j;
      aP0[j] = (bf16)sc[h][m][kk];
      aP1[j] = (bf16)sc[h][m][32 + kk];
    }

    // ---- attn @ V: V^T layout -> B-frags are 16 contiguous bf16 per lane
#pragma unroll
    for (int dt = 0; dt < 4; ++dt) {
      const bf16* vr = vbase + (size_t)(dt * 16 + n) * Sq + k0 + half * 16;
      bf16x16 bv0 = load16(vr);
      bf16x16 bv1 = load16(vr + 32);
      acc[dt] = WMMA_BF16(aP0, bv0, acc[dt]);
      acc[dt] = WMMA_BF16(aP1, bv1, acc[dt]);
    }
    __syncthreads();                                // protect sc[] for next chunk
  }

  // ---- write combined-head output [B,S,D] in bf16 for the final projection
  const size_t orow0 = (size_t)b * Sq + q0;
#pragma unroll
  for (int dt = 0; dt < 4; ++dt) {
#pragma unroll
    for (int r = 0; r < 8; ++r) {
      const int mm = r + 8 * half;
      ob[(orow0 + mm) * Dm + h * 64 + dt * 16 + n] = (bf16)acc[dt][r];
    }
  }
}

// ---------------------------------------------------------------------------
// Output projection + ReLU: out = relu(O @ Wc.T), fp32 result.  Same 32x64
// register blocking as qkv_gemm.  grid = (Mrows/128, Dm/64), block = 128.
// ---------------------------------------------------------------------------
__global__ __launch_bounds__(128) void out_proj_relu(
    const bf16* __restrict__ ob, const bf16* __restrict__ wc, float* __restrict__ out) {
  const int tid = threadIdx.x, wave = tid >> 5, lane = tid & 31;
  const int m = lane & 15, half = lane >> 4, n = m;
  const int row0 = blockIdx.x * 128 + wave * 32;
  const int n0   = blockIdx.y * 64;

  const bf16* arow0 = ob + (size_t)(row0 + m) * Dm;
  const bf16* arow1 = arow0 + (size_t)16 * Dm;
  const bf16* brow  = wc + (size_t)(n0 + n) * Dm;

  f32x8 acc[2][4] = {{{}, {}, {}, {}}, {{}, {}, {}, {}}};
#pragma unroll 4
  for (int kc = 0; kc < Dm / 32; ++kc) {
    const int d0 = kc * 32;
    bf16x16 a0 = load16(arow0 + d0 + half * 16);
    bf16x16 a1 = load16(arow1 + d0 + half * 16);
#pragma unroll
    for (int nt = 0; nt < 4; ++nt) {
      bf16x16 b = load16(brow + (size_t)nt * 16 * Dm + d0 + half * 16);
      acc[0][nt] = WMMA_BF16(a0, b, acc[0][nt]);
      acc[1][nt] = WMMA_BF16(a1, b, acc[1][nt]);
    }
  }
#pragma unroll
  for (int mt = 0; mt < 2; ++mt)
#pragma unroll
    for (int nt = 0; nt < 4; ++nt)
#pragma unroll
      for (int r = 0; r < 8; ++r) {
        const int grow = row0 + mt * 16 + r + 8 * half;
        out[(size_t)grow * Dm + n0 + nt * 16 + n] = fmaxf(acc[mt][nt][r], 0.0f);
      }
}

// ---------------------------------------------------------------------------
// Launch.  Workspace layout (bf16 elements), total ~42 MB:
//   xb[8MB] wq/wk/wv/wc[512KB ea] Q[8MB] K[8MB] V^T[8MB] O[8MB]
// ---------------------------------------------------------------------------
extern "C" void kernel_launch(void* const* d_in, const int* in_sizes, int n_in,
                              void* d_out, int out_size, void* d_ws, size_t ws_size,
                              hipStream_t stream) {
  const float* x  = (const float*)d_in[0];
  const float* Wq = (const float*)d_in[1];
  const float* Wk = (const float*)d_in[2];
  const float* Wv = (const float*)d_in[3];
  const float* Wc = (const float*)d_in[4];

  const size_t XE = (size_t)Bz * Sq * Dm;   // 4,194,304
  const size_t WE = (size_t)Dm * Dm;        //   262,144

  bf16* xb  = (bf16*)d_ws;
  bf16* wqb = xb  + XE;
  bf16* wkb = wqb + WE;
  bf16* wvb = wkb + WE;
  bf16* wcb = wvb + WE;
  bf16* qh  = wcb + WE;
  bf16* kh  = qh  + XE;
  bf16* vt  = kh  + XE;
  bf16* ob  = vt  + XE;

  // 1) fp32 -> bf16 (8 elems/thread)
  cvt_f32_bf16<<<(int)(XE / 8 / 256), 256, 0, stream>>>(x,  xb,  (int)(XE / 8));
  cvt_f32_bf16<<<(int)(WE / 8 / 256), 256, 0, stream>>>(Wq, wqb, (int)(WE / 8));
  cvt_f32_bf16<<<(int)(WE / 8 / 256), 256, 0, stream>>>(Wk, wkb, (int)(WE / 8));
  cvt_f32_bf16<<<(int)(WE / 8 / 256), 256, 0, stream>>>(Wv, wvb, (int)(WE / 8));
  cvt_f32_bf16<<<(int)(WE / 8 / 256), 256, 0, stream>>>(Wc, wcb, (int)(WE / 8));

  // 2) QKV projections (z picks matrix + output layout)
  qkv_gemm<<<dim3(Mrows / 128, Dm / 64, 3), 128, 0, stream>>>(
      xb, wqb, wkb, wvb, qh, kh, vt);

  // 3) fused attention (scores never leave LDS/VGPRs)
  attn_fused<<<dim3(Sq / 16, Bz), 256, 0, stream>>>(qh, kh, vt, ob);

  // 4) output projection + ReLU -> fp32 d_out
  out_proj_relu<<<dim3(Mrows / 128, Dm / 64), 128, 0, stream>>>(
      ob, wcb, (float*)d_out);
}
